// DMSRNN_25898652795354
// MI455X (gfx1250) — matrix operand order repaired and checked by reading, not compiled, not run on hardware
//
#include <hip/hip_runtime.h>

// Problem constants (match reference setup_inputs)
#define ALPHA_C 0.2f
#define Bn 128
#define Tn 1000
#define In 64
#define Hn 512

typedef __attribute__((ext_vector_type(16))) _Float16 v16h;
typedef __attribute__((ext_vector_type(8)))  _Float16 v8h;
typedef __attribute__((ext_vector_type(8)))  float    v8f;

static __device__ __forceinline__ v16h cat8(v8h a, v8h b) {
    return __builtin_shufflevector(a, b, 0,1,2,3,4,5,6,7,8,9,10,11,12,13,14,15);
}

// ---------------------------------------------------------------------------
// Kernel 1: convert W_hh (512x512) and W_in (512x64) to f16 in scratch.
// ---------------------------------------------------------------------------
__global__ __launch_bounds__(256) void k_convert(const float* __restrict__ Whh,
                                                 const float* __restrict__ Win,
                                                 _Float16* __restrict__ Wh16,
                                                 _Float16* __restrict__ Wi16) {
    int i = blockIdx.x * blockDim.x + threadIdx.x;
    if (i < Hn * Hn) Wh16[i] = (_Float16)Whh[i];
    if (i < Hn * In) Wi16[i] = (_Float16)Win[i];
}

// ---------------------------------------------------------------------------
// Kernel 2: xin[t,b,h] = b_in[h] + sum_i x[b,t,i] * W_in[h,i]
// One wave per 16(b) x 16(h) tile, K=64 -> two v_wmma_f32_16x16x32_f16.
// Result written into the d_out "output" region (consumed+overwritten by scan).
// ---------------------------------------------------------------------------
__global__ __launch_bounds__(256) void k_xin(const float* __restrict__ x,
                                             const _Float16* __restrict__ Wi16,
                                             const float* __restrict__ b_in,
                                             float* __restrict__ xin) {
    const int lane  = threadIdx.x & 31;
    const int wid   = threadIdx.x >> 5;
    const int gw    = blockIdx.x * 8 + wid;        // global wave id
    const int ht    = gw & 31;                      // 32 h-tiles
    const int bt    = (gw >> 5) & 7;                // 8 b-tiles
    const int t     = gw >> 8;                      // 0..Tn-1
    const int h0    = ht * 16, b0 = bt * 16;
    const int n     = lane & 15;                    // N index (and A row index)
    const int mhi   = (lane >> 4) << 3;             // C/D: lanes 16-31 hold M+8
    const int khalf = (lane >> 4);                  // A/B lane-half K offset

    const float bias = b_in[h0 + n];
    v8f acc;
#pragma unroll
    for (int v = 0; v < 8; ++v) acc[v] = bias;

    // A row for this lane: x[b0+n, t, :]
    const float* xrow = x + ((size_t)(b0 + n) * Tn + t) * In;

#pragma unroll
    for (int kt = 0; kt < 2; ++kt) {
        // A layout (16-bit, 16x32): lanes 0-15 K=kt*32+{0..15},
        //                           lanes 16-31 K=kt*32+{16..31}
        const int kb = kt * 32 + khalf * 16;
        v8h a0, a1;
#pragma unroll
        for (int j = 0; j < 8; ++j) a0[j] = (_Float16)xrow[kb + j];
#pragma unroll
        for (int j = 0; j < 8; ++j) a1[j] = (_Float16)xrow[kb + 8 + j];
        v16h A = cat8(a0, a1);
        // B layout (32x16): lane holds column n, 16 consecutive K values.
        // B(k,n) = W_in[n,k] -> row-major W_in rows are K-contiguous.
        const int ko = kt * 32 + khalf * 16;
        v16h Bv = *(const v16h*)(Wi16 + (size_t)(h0 + n) * In + ko);
        acc = __builtin_amdgcn_wmma_f32_16x16x32_f16(false, A, false, Bv,
                                                     (short)0, acc, false, false);
    }

    float* orow = xin + ((size_t)t * Bn + b0 + mhi) * Hn + h0 + n;
#pragma unroll
    for (int v = 0; v < 8; ++v) orow[(size_t)v * Hn] = acc[v];
}

// ---------------------------------------------------------------------------
// Kernel 3: sequential scan over T.
// 8 workgroups (16 batch rows each), 512 threads = 16 waves. Wave w owns
// H columns [32w, 32w+32): TWO 16x16 output tiles sharing one A fragment,
// halving LDS A-operand traffic (the per-step bottleneck: was 512 KB/step,
// now 256 KB/step vs ~1K clocks of WMMA work per step per WGP).
// r state: fp32 in VGPRs (C/D layout), f16 shared copy double-buffered in
// LDS. One workgroup barrier per timestep.
// drive = r @ W_hh^T + b_hh + xin_t ;  r' = (1-a) r + a tanh(drive)
// ---------------------------------------------------------------------------
__global__ __launch_bounds__(512) void k_scan(const _Float16* __restrict__ Wh16,
                                              const float* __restrict__ b_hh,
                                              float* __restrict__ hist) {
    __shared__ _Float16 rbuf[2][16 * Hn];          // 2 x 16KB double buffer

    const int lane  = threadIdx.x & 31;
    const int w     = threadIdx.x >> 5;            // 0..15 -> pair of h tiles
    const int h0    = w * 32;                      // first tile columns
    const int h1    = h0 + 16;                     // second tile columns
    const int b0    = blockIdx.x * 16;
    const int n     = lane & 15;
    const int mhi   = (lane >> 4) << 3;
    const int khalf = (lane >> 4);

    // r(0) = 0
    for (int i = threadIdx.x; i < 16 * Hn; i += 512) rbuf[0][i] = (_Float16)0.f;
    __syncthreads();

    const float bhh0 = b_hh[h0 + n];
    const float bhh1 = b_hh[h1 + n];
    v8f rs0, rs1;
#pragma unroll
    for (int v = 0; v < 8; ++v) { rs0[v] = 0.f; rs1[v] = 0.f; }

    // B operands: column h+n of W_hh^T == row h+n of W_hh (K-contiguous)
    const _Float16* wbase0 = Wh16 + (size_t)(h0 + n) * Hn + khalf * 16;
    const _Float16* wbase1 = Wh16 + (size_t)(h1 + n) * Hn + khalf * 16;
    const int arow = n * Hn;                       // A row for this lane

    for (int t = 0; t < Tn; ++t) {
        const int cur = t & 1, nxt = cur ^ 1;

        float* xt0 = hist + ((size_t)t * Bn + b0 + mhi) * Hn + h0 + n;
        float* xt1 = xt0 + 16;
        v8f acc0, acc1;
#pragma unroll
        for (int v = 0; v < 8; ++v) acc0[v] = xt0[(size_t)v * Hn] + bhh0;
#pragma unroll
        for (int v = 0; v < 8; ++v) acc1[v] = xt1[(size_t)v * Hn] + bhh1;

        if (t + 1 < Tn)   // warm next step's xin tile into cache
            __builtin_prefetch(hist + ((size_t)(t + 1) * Bn + b0 + mhi) * Hn + h0 + n, 0, 0);

        const _Float16* ab = &rbuf[cur][0] + arow + khalf * 16;
#pragma unroll 4
        for (int kt = 0; kt < 16; ++kt) {          // K = 512 = 16 x 32
            const _Float16* ap = ab + kt * 32;
            v8h a0lo = *(const v8h*)(ap);          // K {0..7} (+16 for hi lanes)
            v8h a0hi = *(const v8h*)(ap + 8);      // K {8..15} (+16 for hi lanes)
            v16h A   = cat8(a0lo, a0hi);
            v16h Bv0 = *(const v16h*)(wbase0 + kt * 32);
            v16h Bv1 = *(const v16h*)(wbase1 + kt * 32);
            acc0 = __builtin_amdgcn_wmma_f32_16x16x32_f16(false, A, false, Bv0,
                                                          (short)0, acc0, false, false);
            acc1 = __builtin_amdgcn_wmma_f32_16x16x32_f16(false, A, false, Bv1,
                                                          (short)0, acc1, false, false);
        }

#pragma unroll
        for (int v = 0; v < 8; ++v) {
            float r0 = (1.f - ALPHA_C) * rs0[v] + ALPHA_C * tanhf(acc0[v]);
            float r1 = (1.f - ALPHA_C) * rs1[v] + ALPHA_C * tanhf(acc1[v]);
            rs0[v] = r0;
            rs1[v] = r1;
            xt0[(size_t)v * Hn] = r0;                           // output[t,b,h0+n]
            xt1[(size_t)v * Hn] = r1;                           // output[t,b,h1+n]
            rbuf[nxt][(v + mhi) * Hn + h0 + n] = (_Float16)r0;  // shared f16 state
            rbuf[nxt][(v + mhi) * Hn + h1 + n] = (_Float16)r1;
        }
        __syncthreads();   // publish rbuf[nxt] before next step's A reads
    }
}

// ---------------------------------------------------------------------------
// Kernel 4: decision[t,b] = b_out + dot(output[t,b,:], W_out). Wave per row.
// ---------------------------------------------------------------------------
__global__ __launch_bounds__(256) void k_dec(const float* __restrict__ hist,
                                             const float* __restrict__ W_out,
                                             const float* __restrict__ b_out,
                                             float* __restrict__ dv) {
    const int lane = threadIdx.x & 31;
    const int wid  = threadIdx.x >> 5;
    const size_t row = (size_t)blockIdx.x * 8 + wid;   // < Tn*Bn
    const float* r = hist + row * Hn;
    float s = 0.f;
#pragma unroll 4
    for (int j = 0; j < 16; ++j) s += r[lane + 32 * j] * W_out[lane + 32 * j];
#pragma unroll
    for (int off = 16; off; off >>= 1) s += __shfl_xor(s, off, 32);
    if (lane == 0) dv[row] = s + b_out[0];
}

// ---------------------------------------------------------------------------
extern "C" void kernel_launch(void* const* d_in, const int* in_sizes, int n_in,
                              void* d_out, int out_size, void* d_ws, size_t ws_size,
                              hipStream_t stream) {
    (void)in_sizes; (void)n_in; (void)out_size; (void)ws_size;
    const float* x     = (const float*)d_in[0];
    const float* W_in  = (const float*)d_in[1];
    const float* b_in  = (const float*)d_in[2];
    const float* W_hh  = (const float*)d_in[3];
    const float* b_hh  = (const float*)d_in[4];
    const float* W_out = (const float*)d_in[5];
    const float* b_out = (const float*)d_in[6];

    float* out  = (float*)d_out;
    float* dv   = out;                        // decision_variable: Tn*Bn floats
    float* hist = out + (size_t)Tn * Bn;      // output: Tn*Bn*Hn floats

    _Float16* Wh16 = (_Float16*)d_ws;                 // 512 KB
    _Float16* Wi16 = Wh16 + (size_t)Hn * Hn;          // 64 KB

    k_convert<<<(Hn * Hn + 255) / 256, 256, 0, stream>>>(W_hh, W_in, Wh16, Wi16);
    k_xin<<<(Tn * 8 * 32) / 8, 256, 0, stream>>>(x, Wi16, b_in, hist);   // 32000 blocks
    k_scan<<<Bn / 16, 512, 0, stream>>>(Wh16, b_hh, hist);               // 8 blocks
    k_dec<<<(Tn * Bn) / 8, 256, 0, stream>>>(hist, W_out, b_out, dv);    // 16000 blocks
}
